// WindowedMultiHeadAttention_6219112644959
// MI455X (gfx1250) — matrix operand order repaired and checked by reading, not compiled, not run on hardware
//
#include <hip/hip_runtime.h>
#include <hip/hip_bf16.h>

// ---------------------------------------------------------------------------
// WindowedMultiHeadAttention on MI455X (gfx1250, wave32, WMMA 16x16x32 bf16)
//   B=2, N=4096, DIM=1024, HEADS=8, HEAD_DIM=128
//   window sizes per head: {0(global),64,64,128,128,256,256,512}, SHIFT=true
// Pipeline (all stages software-pipelined with double-buffered LDS):
//   1) qkv_gemm : x(8192x1024) @ w_qkv^T -> bf16 Q/K/V [H][B][N][128]
//   2) attn     : flash-style online-softmax attention per head (global or
//                 shifted-window with relative bias + swin mask), bf16 WMMA
//   3) proj_gemm: o(8192x1024 bf16) @ w_proj^T + b_proj -> f32 out
// ---------------------------------------------------------------------------

typedef __attribute__((ext_vector_type(16))) __bf16 v16bf;
typedef __attribute__((ext_vector_type(8)))  __bf16 v8bf;
typedef __attribute__((ext_vector_type(2)))  __bf16 v2bf;
typedef __attribute__((ext_vector_type(8)))  float  v8f;

#define NTOK   4096
#define HD     128
#define SCALE  0.08838834764831845f   // 128^-0.5
#define MASKF  (-100.0f)

__device__ __forceinline__ __bf16 f2bf(float f) { return (__bf16)f; }

__device__ __forceinline__ v8f wmma_bf16(v16bf a, v16bf b, v8f c) {
  return __builtin_amdgcn_wmma_f32_16x16x32_bf16(false, a, false, b, (short)0, c,
                                                 false, false);
}

__device__ __forceinline__ v16bf concat8(v8bf lo, v8bf hi) {
  return __builtin_shufflevector(lo, hi, 0, 1, 2, 3, 4, 5, 6, 7,
                                 8, 9, 10, 11, 12, 13, 14, 15);
}

// A-fragment (16x32): lane<16 -> row=lane, K = {0..15};
//                     lane>=16 -> row=lane-16, K = {16..31}
__device__ __forceinline__ v16bf load_afrag(const __bf16* rowptr, int kb8) {
  v8bf lo = *(const v8bf*)(rowptr + kb8 * 2);
  v8bf hi = *(const v8bf*)(rowptr + kb8 * 2 + 8);
  return concat8(lo, hi);
}

// B-fragment (32x16): lane holds col n, 16 contiguous K at kb16=(lane>=16)*16.
__device__ __forceinline__ v16bf load_bfrag(const __bf16* p) {
  v8bf lo = *(const v8bf*)(p);
  v8bf hi = *(const v8bf*)(p + 8);
  return concat8(lo, hi);
}

__device__ __forceinline__ v8bf cvt8(const float* s) {
  v8bf o;
#pragma unroll
  for (int e = 0; e < 8; ++e) o[e] = f2bf(s[e]);
  return o;
}

// ---------------------------------------------------------------------------
// Kernel 1: QKV projection GEMM.  M=8192, Ncols=3072, K=1024.
// Block 256 thr (8 waves). Block tile 64x128; wave tile 16x64 (4 C-frags).
// Double-buffered LDS, one barrier per K-step, prefetch 3 tiles ahead.
// ---------------------------------------------------------------------------
__global__ __launch_bounds__(256)
void qkv_gemm(const float* __restrict__ X, const float* __restrict__ W,
              __bf16* __restrict__ Q, __bf16* __restrict__ K,
              __bf16* __restrict__ V) {
  __shared__ __align__(32) __bf16 As[2][64 * 32];    // [m][k]
  __shared__ __align__(32) __bf16 Bs[2][128 * 32];   // [n][k]
  const int tid  = threadIdx.x;
  const int wave = tid >> 5, lane = tid & 31;
  const int wm = wave & 3, wn = wave >> 2;
  const int m0 = blockIdx.x * 64;
  const int n0 = blockIdx.y * 128;

  v8f acc[4] = {};
  const int arow = lane & 15;
  const int kb8  = (lane >> 4) << 3;
  const int col  = lane & 15;
  const int kb16 = (lane >> 4) << 4;

  // Staging assignment (per thread)
  const int sar = tid >> 2, sac = (tid & 3) * 8;          // A: row, 8 k's
  const int sbn = tid & 127, sbk = (tid >> 7) * 16;       // B: col, 16 k's
  const float* aptr = X + (size_t)(m0 + sar) * 1024 + sac;
  const float* bptr = W + (size_t)(n0 + sbn) * 1024 + sbk;

  { // prologue: stage tile k0=0 into buffer 0
    *(v8bf*)(As[0] + sar * 32 + sac)     = cvt8(aptr);
    *(v8bf*)(Bs[0] + sbn * 32 + sbk)     = cvt8(bptr);
    *(v8bf*)(Bs[0] + sbn * 32 + sbk + 8) = cvt8(bptr + 8);
  }
  __syncthreads();

  int cur = 0;
  for (int k0 = 0; k0 < 1024; k0 += 32) {
    const bool hasnext = (k0 + 32) < 1024;
    v8bf na = {}, nb0 = {}, nb1 = {};
    if (hasnext) {                       // issue next-tile global loads now
      __builtin_prefetch(aptr + k0 + 96, 0, 1);
      __builtin_prefetch(bptr + k0 + 96, 0, 1);
      na  = cvt8(aptr + k0 + 32);
      nb0 = cvt8(bptr + k0 + 32);
      nb1 = cvt8(bptr + k0 + 40);
    }

    v16bf a = load_afrag(As[cur] + (wm * 16 + arow) * 32, kb8);
    v16bf bf[4];
#pragma unroll
    for (int t = 0; t < 4; ++t)
      bf[t] = load_bfrag(Bs[cur] + (wn * 64 + t * 16 + col) * 32 + kb16);
#pragma unroll
    for (int t = 0; t < 4; ++t) acc[t] = wmma_bf16(a, bf[t], acc[t]);

    if (hasnext) {                       // fill the other buffer
      *(v8bf*)(As[cur ^ 1] + sar * 32 + sac)     = na;
      *(v8bf*)(Bs[cur ^ 1] + sbn * 32 + sbk)     = nb0;
      *(v8bf*)(Bs[cur ^ 1] + sbn * 32 + sbk + 8) = nb1;
    }
    __syncthreads();
    cur ^= 1;
  }

  // Epilogue: scatter into per-head Q/K/V (bf16)
  const int rowb = (lane >> 4) * 8;
#pragma unroll
  for (int t = 0; t < 4; ++t) {
#pragma unroll
    for (int r = 0; r < 8; ++r) {
      int m = m0 + wm * 16 + rowb + r;        // b*4096 + n
      int c = n0 + wn * 64 + t * 16 + col;    // 0..3071
      int b = m >> 12, n = m & 4095;
      int which = c >> 10, rem = c & 1023;
      int head = rem >> 7, d = rem & 127;
      __bf16* dst = (which == 0) ? Q : (which == 1) ? K : V;
      dst[(((size_t)(head * 2 + b)) * NTOK + n) * HD + d] = f2bf(acc[t][r]);
    }
  }
}

// ---------------------------------------------------------------------------
// Kernel 2: fused attention.  grid=(B*N/64, HEADS), block=128 (4 waves).
// Double-buffered K/V chunk stream, one block barrier per 32-key chunk.
// ---------------------------------------------------------------------------
struct HeadParams {
  int ws[8];
  const float* bias[8];
};

__global__ __launch_bounds__(128)
void attn_kernel(const __bf16* __restrict__ Q, const __bf16* __restrict__ K,
                 const __bf16* __restrict__ V, __bf16* __restrict__ O,
                 HeadParams hp) {
  const int head = blockIdx.y;
  const int ws   = hp.ws[head];
  const float* btab = hp.bias[head];
  const int wsz   = (ws == 0) ? NTOK : ws;
  const int shift = (ws == 0) ? 0 : (ws >> 1);

  const int b  = blockIdx.x >> 6;            // batch
  const int q0 = (blockIdx.x & 63) * 64;     // rolled-space query tile start
  const int wave = threadIdx.x >> 5, lane = threadIdx.x & 31;
  const int qrow0 = q0 + wave * 16;
  const int win    = q0 / wsz;
  const int kstart = win * wsz;
  const int kend   = kstart + wsz;
  const bool lastwin = (win == (NTOK / wsz) - 1);

  const size_t hb = (size_t)(head * 2 + b) * NTOK * HD;
  const __bf16* Qh = Q + hb;
  const __bf16* Kh = K + hb;
  const __bf16* Vh = V + hb;

  __shared__ __align__(32) __bf16 Ks[2][32 * HD];   // [key][d]  : S B-frags
  __shared__ __align__(32) __bf16 Vt[2][HD * 32];   // [d][key]  : PV B-frags
  __shared__ __align__(32) __bf16 Ps[4][16 * 32];

  const int arow = lane & 15;
  const int kb8  = (lane >> 4) << 3;
  const int col  = lane & 15;
  const int kb16 = (lane >> 4) << 4;
  const int mb   = (lane >> 4) * 8;          // row base for C-frags

  // Staging assignment (per thread)
  const int skk = threadIdx.x >> 2;                 // K: key row
  const int sdc = (threadIdx.x & 3) * 32;           // K: 32 d's
  const int kk2 = (threadIdx.x & 15) * 2;           // V: key pair
  const int dg  = (threadIdx.x >> 4) * 16;          // V: 16 d's

  // Load Q A-frags (head_dim = GEMM-K = 128 -> 4 frags), rolled -> orig index
  v16bf qa[4];
  {
    int qorig = (qrow0 + arow + shift) & (NTOK - 1);
    const __bf16* s = Qh + (size_t)qorig * HD;
#pragma unroll
    for (int f = 0; f < 4; ++f) qa[f] = load_afrag(s + f * 32, kb8);
  }

  float mrow[8], lrow[8];
  v8f oacc[8] = {};
#pragma unroll
  for (int r = 0; r < 8; ++r) { mrow[r] = -1e30f; lrow[r] = 0.0f; }

  { // prologue: stage chunk 0 into buffer 0
    int korig = (kstart + skk + shift) & (NTOK - 1);
    const v8bf* ksrc = (const v8bf*)(Kh + (size_t)korig * HD + sdc);
    v8bf* kdst = (v8bf*)(Ks[0] + skk * HD + sdc);
#pragma unroll
    for (int e = 0; e < 4; ++e) kdst[e] = ksrc[e];
    int k0o = (kstart + kk2 + shift) & (NTOK - 1);
    int k1o = (kstart + kk2 + 1 + shift) & (NTOK - 1);
    const __bf16* v0 = Vh + (size_t)k0o * HD + dg;
    const __bf16* v1 = Vh + (size_t)k1o * HD + dg;
    v8bf a0 = *(const v8bf*)(v0), a1 = *(const v8bf*)(v0 + 8);
    v8bf c0 = *(const v8bf*)(v1), c1 = *(const v8bf*)(v1 + 8);
#pragma unroll
    for (int e = 0; e < 8; ++e) {
      v2bf p; p[0] = a0[e]; p[1] = c0[e];
      *(v2bf*)(Vt[0] + (dg + e) * 32 + kk2) = p;
    }
#pragma unroll
    for (int e = 0; e < 8; ++e) {
      v2bf p; p[0] = a1[e]; p[1] = c1[e];
      *(v2bf*)(Vt[0] + (dg + 8 + e) * 32 + kk2) = p;
    }
  }
  __syncthreads();

  int cur = 0;
  for (int kc = kstart; kc < kend; kc += 32) {
    const bool hasnext = (kc + 32) < kend;
    v8bf nk[4] = {};
    v8bf nv00 = {}, nv01 = {}, nv10 = {}, nv11 = {};
    if (hasnext) {                       // issue next-chunk global loads now
      int korig = (kc + 32 + skk + shift) & (NTOK - 1);
      const v8bf* ksrc = (const v8bf*)(Kh + (size_t)korig * HD + sdc);
#pragma unroll
      for (int e = 0; e < 4; ++e) nk[e] = ksrc[e];
      int k0o = (kc + 32 + kk2 + shift) & (NTOK - 1);
      int k1o = (kc + 33 + kk2 + shift) & (NTOK - 1);
      const __bf16* v0 = Vh + (size_t)k0o * HD + dg;
      const __bf16* v1 = Vh + (size_t)k1o * HD + dg;
      nv00 = *(const v8bf*)(v0); nv01 = *(const v8bf*)(v0 + 8);
      nv10 = *(const v8bf*)(v1); nv11 = *(const v8bf*)(v1 + 8);
    }

    // S = Q @ K^T for two 16-key sub-tiles (B-frag contiguous in d)
    v8f stile[2];
#pragma unroll
    for (int t2 = 0; t2 < 2; ++t2) {
      v16bf bf[4];
#pragma unroll
      for (int f = 0; f < 4; ++f)
        bf[f] = load_bfrag(Ks[cur] + (t2 * 16 + col) * HD + f * 32 + kb16);
      v8f a = {};
#pragma unroll
      for (int f = 0; f < 4; ++f) a = wmma_bf16(qa[f], bf[f], a);
      stile[t2] = a;
    }

    // scale + bias + shift-mask, then online-softmax update
    float pv[8][2];
#pragma unroll
    for (int r = 0; r < 8; ++r) {
      int m = mb + r;
      int qi = (qrow0 + m) - kstart;               // in-window query idx
      float sv[2];
      float rmax = -1e30f;
#pragma unroll
      for (int t2 = 0; t2 < 2; ++t2) {
        int j = (kc - kstart) + t2 * 16 + col;     // in-window key idx
        float v = stile[t2][r] * SCALE;
        if (btab) v += btab[(qi - j) + wsz - 1];
        if (shift && lastwin) {
          int qp = qrow0 + m, kp = kc + t2 * 16 + col;
          int rq = (qp >= NTOK - shift) ? 2 : (qp >= NTOK - wsz) ? 1 : 0;
          int rk = (kp >= NTOK - shift) ? 2 : (kp >= NTOK - wsz) ? 1 : 0;
          if (rq != rk) v += MASKF;
        }
        sv[t2] = v;
        rmax = fmaxf(rmax, v);
      }
#pragma unroll
      for (int off = 1; off < 16; off <<= 1)       // 16-lane (column) reduce
        rmax = fmaxf(rmax, __shfl_xor(rmax, off, 32));
      float mnew  = fmaxf(mrow[r], rmax);
      float alpha = __expf(mrow[r] - mnew);
      float psum  = 0.0f;
#pragma unroll
      for (int t2 = 0; t2 < 2; ++t2) {
        float p = __expf(sv[t2] - mnew);
        pv[r][t2] = p;
        psum += p;
      }
#pragma unroll
      for (int off = 1; off < 16; off <<= 1)
        psum += __shfl_xor(psum, off, 32);
      lrow[r] = lrow[r] * alpha + psum;
      mrow[r] = mnew;
#pragma unroll
      for (int t = 0; t < 8; ++t) oacc[t][r] *= alpha;
    }

    // P (C-layout) -> wave-private LDS -> A-layout frag (wave-local ordering)
#pragma unroll
    for (int r = 0; r < 8; ++r) {
      int m = mb + r;
      Ps[wave][m * 32 + col]      = f2bf(pv[r][0]);
      Ps[wave][m * 32 + 16 + col] = f2bf(pv[r][1]);
    }
    __builtin_amdgcn_wave_barrier();
    v16bf pa = load_afrag(Ps[wave] + arow * 32, kb8);

    // O = P @ V, 8 col-tiles of 16 (two batches of 4)
#pragma unroll
    for (int g = 0; g < 2; ++g) {
      v16bf bf[4];
#pragma unroll
      for (int t = 0; t < 4; ++t)
        bf[t] = load_bfrag(Vt[cur] + ((g * 4 + t) * 16 + col) * 32 + kb16);
#pragma unroll
      for (int t = 0; t < 4; ++t)
        oacc[g * 4 + t] = wmma_bf16(pa, bf[t], oacc[g * 4 + t]);
    }

    if (hasnext) {                        // fill the other buffer
      v8bf* kdst = (v8bf*)(Ks[cur ^ 1] + skk * HD + sdc);
#pragma unroll
      for (int e = 0; e < 4; ++e) kdst[e] = nk[e];
#pragma unroll
      for (int e = 0; e < 8; ++e) {
        v2bf p; p[0] = nv00[e]; p[1] = nv10[e];
        *(v2bf*)(Vt[cur ^ 1] + (dg + e) * 32 + kk2) = p;
      }
#pragma unroll
      for (int e = 0; e < 8; ++e) {
        v2bf p; p[0] = nv01[e]; p[1] = nv11[e];
        *(v2bf*)(Vt[cur ^ 1] + (dg + 8 + e) * 32 + kk2) = p;
      }
    }
    __syncthreads();
    cur ^= 1;
  }

  // Epilogue: normalize and write bf16 O at [(b,n_orig), head*128 + d]
#pragma unroll
  for (int r = 0; r < 8; ++r) {
    float inv = 1.0f / lrow[r];
    int m = mb + r;
    int qorig = (qrow0 + m + shift) & (NTOK - 1);
    __bf16* dst = O + ((size_t)b * NTOK + qorig) * 1024 + head * HD;
#pragma unroll
    for (int t = 0; t < 8; ++t)
      dst[t * 16 + col] = f2bf(oacc[t][r] * inv);
  }
}

// ---------------------------------------------------------------------------
// Kernel 3: output projection. M=8192, Ncols=1024, K=1024, +bias, f32 out.
// ---------------------------------------------------------------------------
__global__ __launch_bounds__(256)
void proj_gemm(const __bf16* __restrict__ A, const float* __restrict__ W,
               const float* __restrict__ bias, float* __restrict__ Y) {
  __shared__ __align__(32) __bf16 As[2][64 * 32];    // [m][k]
  __shared__ __align__(32) __bf16 Bs[2][128 * 32];   // [n][k]
  const int tid  = threadIdx.x;
  const int wave = tid >> 5, lane = tid & 31;
  const int wm = wave & 3, wn = wave >> 2;
  const int m0 = blockIdx.x * 64;
  const int n0 = blockIdx.y * 128;

  v8f acc[4] = {};
  const int arow = lane & 15;
  const int kb8  = (lane >> 4) << 3;
  const int col  = lane & 15;
  const int kb16 = (lane >> 4) << 4;

  const int sar = tid >> 2, sac = (tid & 3) * 8;
  const int sbn = tid & 127, sbk = (tid >> 7) * 16;
  const __bf16* aptr = A + (size_t)(m0 + sar) * 1024 + sac;
  const float*  bptr = W + (size_t)(n0 + sbn) * 1024 + sbk;

  { // prologue
    *(v8bf*)(As[0] + sar * 32 + sac)     = *(const v8bf*)(aptr);
    *(v8bf*)(Bs[0] + sbn * 32 + sbk)     = cvt8(bptr);
    *(v8bf*)(Bs[0] + sbn * 32 + sbk + 8) = cvt8(bptr + 8);
  }
  __syncthreads();

  int cur = 0;
  for (int k0 = 0; k0 < 1024; k0 += 32) {
    const bool hasnext = (k0 + 32) < 1024;
    v8bf na = {}, nb0 = {}, nb1 = {};
    if (hasnext) {
      __builtin_prefetch(aptr + k0 + 96, 0, 1);
      __builtin_prefetch(bptr + k0 + 96, 0, 1);
      na  = *(const v8bf*)(aptr + k0 + 32);
      nb0 = cvt8(bptr + k0 + 32);
      nb1 = cvt8(bptr + k0 + 40);
    }

    v16bf a = load_afrag(As[cur] + (wm * 16 + arow) * 32, kb8);
    v16bf bf[4];
#pragma unroll
    for (int t = 0; t < 4; ++t)
      bf[t] = load_bfrag(Bs[cur] + (wn * 64 + t * 16 + col) * 32 + kb16);
#pragma unroll
    for (int t = 0; t < 4; ++t) acc[t] = wmma_bf16(a, bf[t], acc[t]);

    if (hasnext) {
      *(v8bf*)(As[cur ^ 1] + sar * 32 + sac)     = na;
      *(v8bf*)(Bs[cur ^ 1] + sbn * 32 + sbk)     = nb0;
      *(v8bf*)(Bs[cur ^ 1] + sbn * 32 + sbk + 8) = nb1;
    }
    __syncthreads();
    cur ^= 1;
  }

  const int rowb = (lane >> 4) * 8;
#pragma unroll
  for (int t = 0; t < 4; ++t) {
#pragma unroll
    for (int r = 0; r < 8; ++r) {
      int m = m0 + wm * 16 + rowb + r;
      int c = n0 + wn * 64 + t * 16 + col;
      Y[(size_t)m * 1024 + c] = acc[t][r] + bias[c];
    }
  }
}

// ---------------------------------------------------------------------------
extern "C" void kernel_launch(void* const* d_in, const int* in_sizes, int n_in,
                              void* d_out, int out_size, void* d_ws,
                              size_t ws_size, hipStream_t stream) {
  const float* x      = (const float*)d_in[0];
  const float* w_qkv  = (const float*)d_in[1];
  const float* w_proj = (const float*)d_in[2];
  const float* b_proj = (const float*)d_in[3];

  // Workspace: q,k,v [8][2][4096][128] bf16 (16 MB each) + o [2][4096][1024]
  char* wsb = (char*)d_ws;
  const size_t seg = (size_t)8 * 2 * 4096 * 128 * 2;   // 16 MB
  __bf16* qb = (__bf16*)(wsb + 0 * seg);
  __bf16* kb = (__bf16*)(wsb + 1 * seg);
  __bf16* vb = (__bf16*)(wsb + 2 * seg);
  __bf16* ob = (__bf16*)(wsb + 3 * seg);

  qkv_gemm<<<dim3(128, 24), 256, 0, stream>>>(x, w_qkv, qb, kb, vb);

  HeadParams hp;
  const int wsizes[8] = {0, 64, 64, 128, 128, 256, 256, 512};
  int ti = 0;
  for (int h = 0; h < 8; ++h) {
    hp.ws[h]   = wsizes[h];
    hp.bias[h] = wsizes[h] ? (const float*)d_in[4 + ti++] : nullptr;
  }
  attn_kernel<<<dim3(128, 8), 128, 0, stream>>>(qb, kb, vb, ob, hp);

  proj_gemm<<<dim3(128, 8), 256, 0, stream>>>(ob, w_proj, b_proj,
                                              (float*)d_out);
}
